// BiLSTM_CRF_31439160607420
// MI455X (gfx1250) — matrix operand order repaired and checked by reading, not compiled, not run on hardware
//
#include <hip/hip_runtime.h>
#include <hip/hip_bf16.h>
#include <stdint.h>

// ---------------------------------------------------------------------------
// BiLSTM-CRF for MI455X (gfx1250, wave32).
//  * All large matmuls run on V_WMMA_F32_16X16X32_BF16 (fp32 accumulate).
//  * Input projections: LDS-staged 64x64-tile GEMM, 4 M-tiles per wave per
//    B-fragment load (high reuse; weights stay resident in the 192MB L2).
//  * Recurrent scans: ONE persistent kernel per scan (4 blocks x 1024 thr =
//    128 waves) with a software grid barrier per timestep; cell state c is
//    kept in registers for the entire 256-step scan.
// ---------------------------------------------------------------------------

constexpr int Bc  = 64;
constexpr int Tc  = 256;
constexpr int Dc  = 300;
constexpr int HDc = 1024;
constexpr int Hc  = 512;
constexpr int Gc  = 4 * Hc;     // 2048
constexpr int Kp0 = 320;        // D padded to a multiple of 32
constexpr int L0c = 13, L0p = 16;
constexpr int L1c = 25, L1p = 32;
constexpr int BT  = Bc * Tc;    // 16384

typedef __attribute__((ext_vector_type(16))) __bf16 v16bf;
typedef __attribute__((ext_vector_type(8)))  __bf16 v8bf;
typedef __attribute__((ext_vector_type(8)))  float  v8f;

// ---- WMMA fragment loaders (wave32 layouts per CDNA5 ISA 7.12.2) ----------

__device__ __forceinline__ v16bf load_a_frag(const __bf16* A, int lda) {
  // A points at (row0, k0); element (m,k) at A[m*lda + k]. 16x32 bf16 tile.
  int lane = threadIdx.x & 31;
  int m = lane & 15, half = lane >> 4;
  v16bf a;
#pragma unroll
  for (int v = 0; v < 8; ++v) {
    int vv = v & 3, hi = v >> 2;
    int k = 16 * hi + 8 * half + 2 * vv;
    a[2 * v]     = A[(size_t)m * lda + k];
    a[2 * v + 1] = A[(size_t)m * lda + k + 1];
  }
  return a;
}

__device__ __forceinline__ v16bf load_b_frag_wT(const __bf16* W, int ldw) {
  // B[k][n] = W[n][k] (weights stored row-major N x K). W points at (n0,k0).
  int lane = threadIdx.x & 31;
  int n = lane & 15, half = lane >> 4;
  v16bf b;
#pragma unroll
  for (int v = 0; v < 8; ++v) {
    int k = 16 * (v >> 2) + 8 * half + 2 * (v & 3);
    b[2 * v]     = W[(size_t)n * ldw + k];
    b[2 * v + 1] = W[(size_t)n * ldw + k + 1];
  }
  return b;
}

__device__ __forceinline__ float sigmoidf_(float x) {
  return 1.0f / (1.0f + __expf(-x));
}

// ---- software grid barrier (4 co-resident blocks) -------------------------

__device__ __forceinline__ void grid_barrier(unsigned* cnt, unsigned* gen,
                                             unsigned nb) {
  __threadfence();              // release: make h stores visible at L2
  __syncthreads();
  if (threadIdx.x == 0) {
    unsigned g = atomicAdd(gen, 0u);
    if (atomicAdd(cnt, 1u) == nb - 1u) {
      atomicExch(cnt, 0u);
      __threadfence();
      atomicAdd(gen, 1u);
    } else {
      while (atomicAdd(gen, 0u) == g) __builtin_amdgcn_s_sleep(1);
    }
  }
  __syncthreads();
  __threadfence();              // acquire: invalidate WGP$/L0 before h reads
}

// ---- small utility kernels -------------------------------------------------

__global__ void zero_kernel(uint32_t* p, int n) {
  int i = blockIdx.x * blockDim.x + threadIdx.x;
  if (i < n) p[i] = 0u;
}

__global__ void lengths_kernel(const int* words, int* lengths) {
  __shared__ int cnt;
  if (threadIdx.x == 0) cnt = 0;
  __syncthreads();
  int b = blockIdx.x;
  int c = (words[b * Tc + threadIdx.x] > 0) ? 1 : 0;
  atomicAdd(&cnt, c);
  __syncthreads();
  if (threadIdx.x == 0) lengths[b] = cnt;
}

__global__ void conv_w_kernel(const float* src, __bf16* dst, int K, int Kp) {
  size_t idx = (size_t)blockIdx.x * blockDim.x + threadIdx.x;
  int k = (int)(idx % Kp);
  size_t n = idx / Kp;
  dst[idx] = (__bf16)((k < K) ? src[n * K + k] : 0.0f);
}

__global__ void gather_embed_kernel(const int* words, const float* emb,
                                    const int* lengths, __bf16* dst, int rev) {
  size_t idx = (size_t)blockIdx.x * blockDim.x + threadIdx.x;  // B*T*Kp0
  int k = (int)(idx % Kp0);
  size_t bt = idx / Kp0;
  int b = (int)(bt / Tc), t = (int)(bt % Tc);
  float v = 0.0f;
  int st = t;
  bool valid = true;
  if (rev) { st = lengths[b] - 1 - t; valid = (st >= 0); }
  if (valid && k < Dc) {
    int w = words[b * Tc + st];
    v = emb[(size_t)w * Dc + k];
  }
  dst[idx] = (__bf16)v;
}

__global__ void pack_bf16_kernel(const float* src, const int* lengths,
                                 __bf16* dst, int rev) {
  size_t idx = (size_t)blockIdx.x * blockDim.x + threadIdx.x;  // B*T*HD
  int k = (int)(idx % HDc);
  size_t bt = idx / HDc;
  int b = (int)(bt / Tc), t = (int)(bt % Tc);
  float v = 0.0f;
  int st = t;
  bool valid = true;
  if (rev) { st = lengths[b] - 1 - t; valid = (st >= 0); }
  if (valid) v = src[((size_t)b * Tc + st) * HDc + k];
  dst[idx] = (__bf16)v;
}

__global__ void unrev_store_kernel(const float* ystmp, const int* lengths,
                                   float* out) {
  size_t idx = (size_t)blockIdx.x * blockDim.x + threadIdx.x;  // B*T*H
  int c = (int)(idx % Hc);
  size_t bt = idx / Hc;
  int b = (int)(bt / Tc), t = (int)(bt % Tc);
  int st = lengths[b] - 1 - t;
  float v = (st >= 0) ? ystmp[((size_t)b * Tc + st) * Hc + c] : 0.0f;
  out[((size_t)b * Tc + t) * HDc + Hc + c] = v;
}

// ---- WMMA GEMM: C[M x 2048] = A[M x K](bf16) * W[2048 x K]^T + bias --------
// grid = (M/64, N/64), block = 128 (4 waves). A 64x32 K-slice is staged in
// LDS once per block; each wave register-blocks 4 M-tiles per B-fragment.

__global__ __launch_bounds__(128)
void gemm_bias_kernel(const __bf16* __restrict__ A, const __bf16* __restrict__ W,
                      const float* __restrict__ bias, float* __restrict__ C,
                      int K, int lda, int ldw) {
  __shared__ __bf16 As[64 * 32];
  int wid  = threadIdx.x >> 5;
  int row0 = blockIdx.x * 64;
  int col0 = blockIdx.y * 64 + wid * 16;
  int lane = threadIdx.x & 31;
  int half = lane >> 4, nn = lane & 15;

  v8f acc[4];
  float bv = bias[col0 + nn];
#pragma unroll
  for (int mt = 0; mt < 4; ++mt)
#pragma unroll
    for (int r = 0; r < 8; ++r) acc[mt][r] = bv;

  int lr = threadIdx.x >> 1;          // 0..63  : A row within tile
  int lc = (threadIdx.x & 1) * 16;    // 0 / 16 : A col half

  for (int k = 0; k < K; k += 32) {
    __syncthreads();
    const __bf16* Ag = A + (size_t)(row0 + lr) * lda + k + lc;
    v8bf t0 = *(const v8bf*)(Ag);
    v8bf t1 = *(const v8bf*)(Ag + 8);
    *(v8bf*)(&As[lr * 32 + lc])     = t0;
    *(v8bf*)(&As[lr * 32 + lc + 8]) = t1;
    if (k + 32 < K)  // pull next B K-slice toward the caches
      __builtin_prefetch(W + (size_t)(col0 + nn) * ldw + k + 32, 0, 1);
    __syncthreads();
    v16bf b = load_b_frag_wT(W + (size_t)col0 * ldw + k, ldw);
#pragma unroll
    for (int mt = 0; mt < 4; ++mt) {
      v16bf a = load_a_frag(&As[mt * 16 * 32], 32);
      acc[mt] = __builtin_amdgcn_wmma_f32_16x16x32_bf16(false, a, false, b,
                                                        (short)0, acc[mt],
                                                        false, false);
    }
  }
#pragma unroll
  for (int mt = 0; mt < 4; ++mt)
#pragma unroll
    for (int r = 0; r < 8; ++r) {
      int m = mt * 16 + r + 8 * half;
      C[((size_t)(row0 + m)) * Gc + col0 + nn] = acc[mt][r];
    }
}

// ---- persistent fused LSTM scan -------------------------------------------
// 4 blocks x 1024 threads = 128 waves; wave w owns batch-tile (w>>5) and
// H-column-tile (w&31). Runs all 256 timesteps with a grid barrier between
// steps. Cell state c lives in registers for the whole scan.

__global__ __launch_bounds__(1024)
void lstm_scan_kernel(const float* __restrict__ xp, const __bf16* __restrict__ Whh,
                      const int* __restrict__ words, __bf16* __restrict__ hb,
                      float* __restrict__ ys, int ldys, int ycol0,
                      unsigned* __restrict__ bar) {
  int gwave = blockIdx.x * (blockDim.x >> 5) + (threadIdx.x >> 5);
  int n0 = (gwave & 31) * 16;   // column within H
  int b0 = (gwave >> 5) * 16;   // batch row tile
  int lane = threadIdx.x & 31;
  int half = lane >> 4, nn = lane & 15;
  unsigned* cnt = bar;
  unsigned* gen = bar + 1;

  float creg[8];
#pragma unroll
  for (int r = 0; r < 8; ++r) creg[r] = 0.0f;

  // zero h(0): each wave zeroes exactly its own (b,col) tile of hb[0]
#pragma unroll
  for (int r = 0; r < 8; ++r) {
    int b = b0 + r + 8 * half;
    hb[(size_t)b * Hc + n0 + nn] = (__bf16)0.0f;
  }
  grid_barrier(cnt, gen, 4u);

  for (int t = 0; t < Tc; ++t) {
    const __bf16* h_in  = hb + (size_t)(t & 1) * Bc * Hc;
    __bf16*       h_out = hb + (size_t)((t + 1) & 1) * Bc * Hc;

    v8f acc[4];
#pragma unroll
    for (int g = 0; g < 4; ++g)
#pragma unroll
      for (int r = 0; r < 8; ++r) {
        int b = b0 + r + 8 * half;
        acc[g][r] = xp[((size_t)b * Tc + t) * Gc + g * Hc + n0 + nn];
      }

    for (int k = 0; k < Hc; k += 32) {
      v16bf a = load_a_frag(h_in + (size_t)b0 * Hc + k, Hc);
#pragma unroll
      for (int g = 0; g < 4; ++g) {
        v16bf bf = load_b_frag_wT(Whh + (size_t)(g * Hc + n0) * Hc + k, Hc);
        acc[g] = __builtin_amdgcn_wmma_f32_16x16x32_bf16(false, a, false, bf,
                                                         (short)0, acc[g],
                                                         false, false);
      }
    }

#pragma unroll
    for (int r = 0; r < 8; ++r) {
      int b   = b0 + r + 8 * half;
      int col = n0 + nn;
      bool m  = words[b * Tc + t] > 0;
      float iv = sigmoidf_(acc[0][r]);
      float fv = sigmoidf_(acc[1][r]);
      float gv = tanhf(acc[2][r]);
      float ov = sigmoidf_(acc[3][r]);
      float c_new = fv * creg[r] + iv * gv;
      float h_new = ov * tanhf(c_new);
      float h_old = (float)h_in[(size_t)b * Hc + col];
      creg[r] = m ? c_new : creg[r];
      h_out[(size_t)b * Hc + col] = (__bf16)(m ? h_new : h_old);
      ys[((size_t)b * Tc + t) * ldys + ycol0 + col] = m ? h_new : 0.0f;
    }
    grid_barrier(cnt, gen, 4u);
  }
}

// ---- heads ----------------------------------------------------------------

__global__ void head_kernel(const float* __restrict__ feat, const float* __restrict__ w,
                            const float* __restrict__ bias, const int* __restrict__ tags_ext,
                            float* __restrict__ em, int L, int Lp, int ldw) {
  int idx = blockIdx.x * blockDim.x + threadIdx.x;  // BT * Lp
  int l  = idx % Lp;
  int bt = idx / Lp;
  if (l >= L) return;
  const float* f  = feat + (size_t)bt * HDc;
  const float* wr = w + (size_t)l * ldw;
  float s = bias[l];
  for (int k = 0; k < HDc; ++k) s += f[k] * wr[k];
  if (tags_ext) s += (float)tags_ext[bt] * wr[HDc];
  em[(size_t)bt * Lp + l] = s;
}

// ---- Viterbi: one block (32 threads) per batch row ------------------------

__global__ void viterbi_kernel(const float* __restrict__ em, int ldem, int L,
                               const float* __restrict__ start, const float* __restrict__ endv,
                               const float* __restrict__ trans, const int* __restrict__ words,
                               int* __restrict__ hist, int* __restrict__ tags) {
  int b = blockIdx.x;
  int l = threadIdx.x;
  __shared__ float sc[32];
  __shared__ float ns[32];
  if (l < L) sc[l] = start[l] + em[((size_t)b * Tc) * ldem + l];
  __syncthreads();
  for (int t = 1; t < Tc; ++t) {
    float best = -3.4e38f;
    int bi = 0;
    if (l < L) {
      for (int p = 0; p < L; ++p) {
        float v = sc[p] + trans[p * L + l];
        if (v > best) { best = v; bi = p; }
      }
    }
    bool m = words[b * Tc + t] > 0;
    if (l < L) {
      hist[((size_t)b * Tc + t) * 32 + l] = m ? bi : l;
      ns[l] = m ? (best + em[((size_t)b * Tc + t) * ldem + l]) : sc[l];
    }
    __syncthreads();
    if (l < L) sc[l] = ns[l];
    __syncthreads();
  }
  if (l == 0) {
    float bb = -3.4e38f;
    int last = 0;
    for (int p = 0; p < L; ++p) {
      float v = sc[p] + endv[p];
      if (v > bb) { bb = v; last = p; }
    }
    int tag = last;
    tags[b * Tc + Tc - 1] = (words[b * Tc + Tc - 1] > 0) ? tag : 0;
    for (int t = Tc - 2; t >= 0; --t) {
      tag = hist[((size_t)b * Tc + t + 1) * 32 + tag];
      tags[b * Tc + t] = (words[b * Tc + t] > 0) ? tag : 0;
    }
  }
}

__global__ void write_out_kernel(const int* tg0, const int* tg1, const int* lengths,
                                 float* out) {
  int i = blockIdx.x * blockDim.x + threadIdx.x;
  if (i < BT) {
    out[i]      = (float)tg0[i];
    out[BT + i] = (float)tg1[i];
  }
  if (i < Bc) out[2 * BT + i] = (float)lengths[i];
}

// ---------------------------------------------------------------------------

extern "C" void kernel_launch(void* const* d_in, const int* in_sizes, int n_in,
                              void* d_out, int out_size, void* d_ws, size_t ws_size,
                              hipStream_t stream) {
  (void)in_sizes; (void)n_in; (void)out_size; (void)ws_size;

  const int*   words  = (const int*)d_in[0];
  const float* emb    = (const float*)d_in[1];
  const float* l0_wih = (const float*)d_in[2];
  const float* l0_whh = (const float*)d_in[3];
  const float* l0_b   = (const float*)d_in[4];
  const float* l1_wih = (const float*)d_in[5];
  const float* l1_whh = (const float*)d_in[6];
  const float* l1_b   = (const float*)d_in[7];
  const float* h0_w   = (const float*)d_in[8];
  const float* h0_b   = (const float*)d_in[9];
  const float* h1_w   = (const float*)d_in[10];
  const float* h1_b   = (const float*)d_in[11];
  const float* crf0_s = (const float*)d_in[12];
  const float* crf0_e = (const float*)d_in[13];
  const float* crf0_t = (const float*)d_in[14];
  const float* crf1_s = (const float*)d_in[15];
  const float* crf1_e = (const float*)d_in[16];
  const float* crf1_t = (const float*)d_in[17];
  float* out = (float*)d_out;

  uint8_t* ws = (uint8_t*)d_ws;
  size_t off = 0;
  auto alloc = [&](size_t bytes) -> void* {
    void* p = ws + off;
    off = (off + bytes + 255) & ~(size_t)255;
    return p;
  };

  int*      lengths = (int*)alloc(Bc * sizeof(int));
  unsigned* bar     = (unsigned*)alloc(2 * sizeof(unsigned));
  __bf16*   inbf    = (__bf16*)alloc((size_t)BT * HDc * 2);
  __bf16*   w0ih    = (__bf16*)alloc((size_t)2 * Gc * Kp0 * 2);
  __bf16*   w0hh    = (__bf16*)alloc((size_t)2 * Gc * Hc * 2);
  __bf16*   w1ih    = (__bf16*)alloc((size_t)2 * Gc * HDc * 2);
  __bf16*   w1hh    = (__bf16*)alloc((size_t)2 * Gc * Hc * 2);
  float*    xp      = (float*)alloc((size_t)BT * Gc * 4);     // 134 MB
  __bf16*   hb      = (__bf16*)alloc((size_t)2 * Bc * Hc * 2);
  float*    ystmp   = (float*)alloc((size_t)BT * Hc * 4);
  float*    out0    = (float*)alloc((size_t)BT * HDc * 4);
  float*    out1    = (float*)alloc((size_t)BT * HDc * 4);
  float*    em0     = (float*)alloc((size_t)BT * L0p * 4);
  float*    em1     = (float*)alloc((size_t)BT * L1p * 4);
  int*      tg0     = (int*)alloc((size_t)BT * 4);
  int*      tg1     = (int*)alloc((size_t)BT * 4);
  int*      hist    = (int*)alloc((size_t)BT * 32 * 4);

  zero_kernel<<<1, 256, 0, stream>>>((uint32_t*)bar, 2);
  lengths_kernel<<<Bc, Tc, 0, stream>>>(words, lengths);

  for (int d = 0; d < 2; ++d) {
    conv_w_kernel<<<(Gc * Kp0) / 256, 256, 0, stream>>>(
        l0_wih + (size_t)d * Gc * Dc, w0ih + (size_t)d * Gc * Kp0, Dc, Kp0);
    conv_w_kernel<<<(Gc * Hc) / 256, 256, 0, stream>>>(
        l0_whh + (size_t)d * Gc * Hc, w0hh + (size_t)d * Gc * Hc, Hc, Hc);
    conv_w_kernel<<<(Gc * HDc) / 256, 256, 0, stream>>>(
        l1_wih + (size_t)d * Gc * HDc, w1ih + (size_t)d * Gc * HDc, HDc, HDc);
    conv_w_kernel<<<(Gc * Hc) / 256, 256, 0, stream>>>(
        l1_whh + (size_t)d * Gc * Hc, w1hh + (size_t)d * Gc * Hc, Hc, Hc);
  }

  dim3 gemm_grid(BT / 64, Gc / 64);

  auto run_scan = [&](const __bf16* whh, float* ysout, int ldys, int ycol0) {
    lstm_scan_kernel<<<4, 1024, 0, stream>>>(xp, whh, words, hb, ysout, ldys,
                                             ycol0, bar);
  };

  // ---- layer 0 ----
  gather_embed_kernel<<<(BT * Kp0) / 256, 256, 0, stream>>>(words, emb, lengths, inbf, 0);
  gemm_bias_kernel<<<gemm_grid, 128, 0, stream>>>(inbf, w0ih, l0_b, xp, Kp0, Kp0, Kp0);
  run_scan(w0hh, out0, HDc, 0);

  gather_embed_kernel<<<(BT * Kp0) / 256, 256, 0, stream>>>(words, emb, lengths, inbf, 1);
  gemm_bias_kernel<<<gemm_grid, 128, 0, stream>>>(inbf, w0ih + (size_t)Gc * Kp0,
                                                  l0_b + Gc, xp, Kp0, Kp0, Kp0);
  run_scan(w0hh + (size_t)Gc * Hc, ystmp, Hc, 0);
  unrev_store_kernel<<<(BT * Hc) / 256, 256, 0, stream>>>(ystmp, lengths, out0);

  // ---- layer 1 ----
  pack_bf16_kernel<<<(BT * HDc) / 256, 256, 0, stream>>>(out0, lengths, inbf, 0);
  gemm_bias_kernel<<<gemm_grid, 128, 0, stream>>>(inbf, w1ih, l1_b, xp, HDc, HDc, HDc);
  run_scan(w1hh, out1, HDc, 0);

  pack_bf16_kernel<<<(BT * HDc) / 256, 256, 0, stream>>>(out0, lengths, inbf, 1);
  gemm_bias_kernel<<<gemm_grid, 128, 0, stream>>>(inbf, w1ih + (size_t)Gc * HDc,
                                                  l1_b + Gc, xp, HDc, HDc, HDc);
  run_scan(w1hh + (size_t)Gc * Hc, ystmp, Hc, 0);
  unrev_store_kernel<<<(BT * Hc) / 256, 256, 0, stream>>>(ystmp, lengths, out1);

  // ---- heads + CRF ----
  head_kernel<<<(BT * L0p) / 256, 256, 0, stream>>>(out1, h0_w, h0_b, nullptr, em0,
                                                    L0c, L0p, HDc);
  viterbi_kernel<<<Bc, 32, 0, stream>>>(em0, L0p, L0c, crf0_s, crf0_e, crf0_t,
                                        words, hist, tg0);
  head_kernel<<<(BT * L1p) / 256, 256, 0, stream>>>(out1, h1_w, h1_b, tg0, em1,
                                                    L1c, L1p, HDc + 1);
  viterbi_kernel<<<Bc, 32, 0, stream>>>(em1, L1p, L1c, crf1_s, crf1_e, crf1_t,
                                        words, hist, tg1);

  write_out_kernel<<<(BT + 255) / 256, 256, 0, stream>>>(tg0, tg1, lengths, out);
}
